// PoolingAttention_25366076850335
// MI455X (gfx1250) — matrix operand order repaired and checked by reading, not compiled, not run on hardware
//
#include <hip/hip_runtime.h>
#include <math.h>

typedef __attribute__((ext_vector_type(16))) _Float16 v16h;
typedef __attribute__((ext_vector_type(8)))  float    v8f;
typedef __attribute__((ext_vector_type(4)))  unsigned int u32x4;
typedef __attribute__((ext_vector_type(4)))  int      i32x4;
typedef __attribute__((ext_vector_type(8)))  int      i32x8;

#define BATCH  16
#define C_CH   128
#define HH     32
#define WW     32
#define NPIX   1024
#define HEADS  4
#define DHEAD  32
#define LVALID 1426
#define LPAD   1472            // pad to multiple of 64 for 64-wide GEMM tiles
#define ATT_SCALE 0.17677669529663687f  // 32^-0.5

#ifndef __has_builtin
#define __has_builtin(x) 0
#endif
#if __has_builtin(__builtin_amdgcn_tensor_load_to_lds) && \
    __has_builtin(__builtin_amdgcn_s_wait_tensorcnt)
#define HAVE_TDM 1
#else
#define HAVE_TDM 0
#endif

// ---------------------------------------------------------------- helpers ---
__device__ __forceinline__ float gelu_exact(float x) {
    return 0.5f * x * (1.0f + erff(x * 0.70710678118654752f));
}
__device__ __forceinline__ float sigmoidf(float x) {
    return 1.0f / (1.0f + __expf(-x));
}
// 16-bit A-matrix (16x32) K index for packed pair p (0..7), half hi (0/1)
__device__ __forceinline__ int a_kmap(int p, int hi) {
    int base = 2 * p;
    return base + (hi ? 16 : 0);
}
// half-wave (16-lane) reductions; masks <16 never cross the 16-lane boundary
__device__ __forceinline__ float halfmax(float v) {
    for (int o = 1; o < 16; o <<= 1) v = fmaxf(v, __shfl_xor(v, o, 32));
    return v;
}
__device__ __forceinline__ float halfsum(float v) {
    for (int o = 1; o < 16; o <<= 1) v += __shfl_xor(v, o, 32);
    return v;
}

#if HAVE_TDM
// Issue a TDM 2-D tile load: tile_h rows of tile_w fp32 elements, global row
// pitch = row_stride elements, packed contiguously into LDS at lds_off.
// D# layout per CDNA5 ISA (group0 128b, group1 256b).
__device__ __forceinline__ void tdm_load_2d(unsigned lds_off, const void* gptr,
                                            int tile_w, int tile_h,
                                            int row_stride) {
    unsigned long long ga = (unsigned long long)gptr;
    u32x4 g0;
    g0[0] = 1u;                                   // count=1 valid descriptor
    g0[1] = lds_off;                              // lds_addr (bytes)
    g0[2] = (unsigned)(ga & 0xffffffffu);         // global_addr[31:0]
    g0[3] = (unsigned)((ga >> 32) & 0x01ffffffu)  // global_addr[56:32]
            | (2u << 30);                         // type=2 ("image")
    i32x8 g1;
    g1[0] = (2 << 16);                            // data_size = 4 bytes
    g1[1] = (tile_w & 0xffff) << 16;              // tensor_dim0[15:0]
    g1[2] = ((unsigned)tile_w >> 16) |            // tensor_dim0[31:16]
            ((tile_h & 0xffff) << 16);            // tensor_dim1[15:0]
    g1[3] = ((unsigned)tile_h >> 16) |            // tensor_dim1[31:16]
            ((tile_w & 0xffff) << 16);            // tile_dim0
    g1[4] = (tile_h & 0xffff);                    // tile_dim1 (tile_dim2=0)
    g1[5] = row_stride;                           // tensor_dim0_stride[31:0]
    g1[6] = 0;                                    // stride hi / dim1_stride lo
    g1[7] = 0;
    i32x4 z4 = {0, 0, 0, 0};
#if __clang_major__ >= 23
    i32x8 z8 = {0, 0, 0, 0, 0, 0, 0, 0};
    __builtin_amdgcn_tensor_load_to_lds(g0, g1, z4, z4, z8, 0);
#else
    __builtin_amdgcn_tensor_load_to_lds(g0, g1, z4, z4, 0);
#endif
}
#endif

// ------------------------------------------------------------- transpose ---
// x (B,N,C) -> x_img (B,C,N)
__global__ void k_transpose(const float* __restrict__ x, float* __restrict__ xi) {
    int idx = blockIdx.x * blockDim.x + threadIdx.x;
    if (idx >= BATCH * NPIX * C_CH) return;
    int c = idx % C_CH;
    int t = idx / C_CH;
    int n = t % NPIX;
    int b = t / NPIX;
    xi[(b * C_CH + c) * NPIX + n] = x[idx];
}

// --------------------------------------------------- generic depthwise conv ---
__global__ void k_dwconv(const float* __restrict__ in, const float* __restrict__ w,
                         const float* __restrict__ bias, float* __restrict__ out,
                         int Hh, int Ww, int kh, int kw, int ph, int pw,
                         int dh, int dw, int gelu_in) {
    int idx = blockIdx.x * blockDim.x + threadIdx.x;
    int total = BATCH * C_CH * Hh * Ww;
    if (idx >= total) return;
    int x = idx % Ww;
    int t = idx / Ww;
    int y = t % Hh;
    t /= Hh;
    int c = t % C_CH;
    int b = t / C_CH;
    const float* ip = in + ((long)(b * C_CH + c)) * Hh * Ww;
    const float* wp = w + c * kh * kw;
    float s = bias[c];
    for (int ky = 0; ky < kh; ++ky) {
        int yy = y - ph + ky * dh;
        if ((unsigned)yy >= (unsigned)Hh) continue;
        for (int kx = 0; kx < kw; ++kx) {
            int xx = x - pw + kx * dw;
            if ((unsigned)xx >= (unsigned)Ww) continue;
            float v = ip[yy * Ww + xx];
            if (gelu_in) v = gelu_exact(v);
            s += wp[ky * kw + kx] * v;
        }
    }
    out[idx] = s;
}

// -------------------------------------------------------------- WMMA GEMM ---
// D[co,n] = sum_k Wmat[co,k] * Bsrc[b][k,n] (+bias).
// 4-wave workgroup computes a 64(co) x 64(n) tile. Per 32-K step, wave 0
// stages A (64x32) and B (32x64) slabs into LDS via TDM tensor loads
// (fallback: cooperative copies); each wave then runs 4 WMMA from LDS.
// out_nc==0: Out[b*oStrideB + co*Nn + n];  out_nc==1: Out[b*oStrideB + n*M + co]
__global__ __launch_bounds__(128)
void k_gemm_wmma(const float* __restrict__ Wmat, const float* __restrict__ bias,
                 const float* __restrict__ Bsrc, float* __restrict__ Out,
                 int M, int Nn, int K, long bStrideB, int ldb,
                 long oStrideB, int out_nc) {
    __shared__ float ldsA[64][32];   // [co row][k]
    __shared__ float ldsB[32][64];   // [k][n]
    int tid = threadIdx.x;
    int wv = tid >> 5, lane = tid & 31;
    int hi = lane >> 4, lr = lane & 15;
    int n0 = blockIdx.x * 64, coB = blockIdx.y * 64, b = blockIdx.z;
    const float* Bb = Bsrc + (long)b * bStrideB;

    v8f zero8 = {0.f, 0.f, 0.f, 0.f, 0.f, 0.f, 0.f, 0.f};
    v8f acc[4];
    acc[0] = zero8; acc[1] = zero8; acc[2] = zero8; acc[3] = zero8;

    for (int k0 = 0; k0 < K; k0 += 32) {
        __syncthreads();   // previous-iteration LDS reads complete
#if HAVE_TDM
        if (wv == 0) {
            tdm_load_2d((unsigned)(unsigned long long)(void*)&ldsA[0][0],
                        Wmat + (long)coB * K + k0, 32, 64, K);
            tdm_load_2d((unsigned)(unsigned long long)(void*)&ldsB[0][0],
                        Bb + (long)k0 * ldb + n0, 64, 32, ldb);
            __builtin_amdgcn_s_wait_tensorcnt(0);
        }
        asm volatile("" ::: "memory");
#else
        for (int i = tid; i < 64 * 32; i += 128) {
            int r = i >> 5, kk = i & 31;
            ldsA[r][kk] = Wmat[(long)(coB + r) * K + k0 + kk];
        }
        for (int i = tid; i < 32 * 64; i += 128) {
            int kk = i >> 6, nn = i & 63;
            ldsB[kk][nn] = Bb[(long)(k0 + kk) * ldb + n0 + nn];
        }
#endif
        __syncthreads();
        v16h af;
#pragma unroll
        for (int j = 0; j < 16; ++j) {
            int k = a_kmap(j >> 1, hi) + (j & 1);
            af[j] = (_Float16)ldsA[wv * 16 + lr][k];
        }
#pragma unroll
        for (int nt = 0; nt < 4; ++nt) {
            v16h bf;
#pragma unroll
            for (int j = 0; j < 8; ++j) {
                int kd = (hi ? 16 : 0) + 2 * j;
                bf[2 * j]     = (_Float16)ldsB[kd][nt * 16 + lr];
                bf[2 * j + 1] = (_Float16)ldsB[kd + 1][nt * 16 + lr];
            }
            acc[nt] = __builtin_amdgcn_wmma_f32_16x16x32_f16(
                false, af, false, bf, (short)0, acc[nt], false, false);
        }
    }
#pragma unroll
    for (int nt = 0; nt < 4; ++nt) {
#pragma unroll
        for (int r = 0; r < 8; ++r) {
            int co = coB + wv * 16 + r + hi * 8;
            int n = n0 + nt * 16 + lr;
            float v = acc[nt][r] + (bias ? bias[co] : 0.f);
            if (out_nc) Out[(long)b * oStrideB + (long)n * M + co] = v;
            else        Out[(long)b * oStrideB + (long)co * Nn + n] = v;
        }
    }
}

// -------------------------------------------- conv3 3x3 C->C implicit GEMM ---
// x3[b,co,n] = sum_{g=(ci,ky,kx)} W3[co,g] * im2col(x2)[g,n],  K = 128*9
__global__ __launch_bounds__(32)
void k_conv3_wmma(const float* __restrict__ W3, const float* __restrict__ bias,
                  const float* __restrict__ x2, float* __restrict__ Out) {
    int lane = threadIdx.x & 31;
    int hi = lane >> 4, lr = lane & 15;
    int n0 = blockIdx.x * 16, co0 = blockIdx.y * 16, b = blockIdx.z;
    int n = n0 + lr, y = n >> 5, x = n & 31;
    const float* x2b = x2 + (long)b * C_CH * NPIX;
    v8f acc = {0.f, 0.f, 0.f, 0.f, 0.f, 0.f, 0.f, 0.f};
    for (int k0 = 0; k0 < 1152; k0 += 32) {
        v16h af, bf;
#pragma unroll
        for (int j = 0; j < 16; ++j) {
            int g = k0 + a_kmap(j >> 1, hi) + (j & 1);
            af[j] = (_Float16)W3[(long)(co0 + lr) * 1152 + g];
        }
#pragma unroll
        for (int j = 0; j < 8; ++j) {
            int kb = (hi ? 16 : 0) + 2 * j;
#pragma unroll
            for (int e = 0; e < 2; ++e) {
                int g = k0 + kb + e;
                int ci = g / 9, tap = g - ci * 9;
                int ky = tap / 3 - 1, kx = tap - (tap / 3) * 3 - 1;
                int yy = y + ky, xx = x + kx;
                float v = 0.f;
                if ((unsigned)yy < 32u && (unsigned)xx < 32u)
                    v = x2b[(ci << 10) + (yy << 5) + xx];
                bf[2 * j + e] = (_Float16)v;
            }
        }
        acc = __builtin_amdgcn_wmma_f32_16x16x32_f16(false, af, false, bf,
                                                     (short)0, acc, false, false);
    }
#pragma unroll
    for (int r = 0; r < 8; ++r) {
        int co = co0 + r + hi * 8;
        Out[((long)b * C_CH + co) * NPIX + n] = acc[r] + bias[co];
    }
}

// ------------------------------------------------------------ elementwise ---
__global__ void k_mul(const float* a, const float* b, float* o, int total) {
    int i = blockIdx.x * blockDim.x + threadIdx.x;
    if (i < total) o[i] = a[i] * b[i];
}
__global__ void k_add_inplace(float* a, const float* b, int total) {
    int i = blockIdx.x * blockDim.x + threadIdx.x;
    if (i < total) a[i] += b[i];
}
__global__ void k_gate_ca(const float* __restrict__ xi, const float* __restrict__ ca,
                          float* __restrict__ o) {
    int i = blockIdx.x * blockDim.x + threadIdx.x;
    if (i >= BATCH * C_CH * NPIX) return;
    int bc = i / NPIX;
    o[i] = ca[bc] * xi[i];
}

// ------------------------------------------------------- channel attention ---
__global__ void k_chanatt(const float* __restrict__ xi,
                          const float* a1w, const float* a1b,
                          const float* a2w, const float* a2b,
                          const float* m1w, const float* m1b,
                          const float* m2w, const float* m2b,
                          float* __restrict__ ca) {
    __shared__ float avg[C_CH], mxs[C_CH], ha[4], hm[4];
    int c = threadIdx.x, b = blockIdx.x;
    const float* p = xi + ((long)b * C_CH + c) * NPIX;
    float s = 0.f, m = -1e30f;
    for (int i = 0; i < NPIX; ++i) { float v = p[i]; s += v; m = fmaxf(m, v); }
    avg[c] = s * (1.0f / NPIX);
    mxs[c] = m;
    __syncthreads();
    if (c < 4) {
        float sa = a1b[c], sm = m1b[c];
        for (int ci = 0; ci < C_CH; ++ci) {
            // 3x3 conv on a zero-padded 1x1 map only sees the center tap
            sa += a1w[(c * C_CH + ci) * 9 + 4] * avg[ci];
            sm += m1w[c * C_CH + ci] * mxs[ci];
        }
        ha[c] = fmaxf(sa, 0.f);
        hm[c] = fmaxf(sm, 0.f);
    }
    __syncthreads();
    float y1 = a2b[c], y2 = m2b[c];
    for (int r = 0; r < 4; ++r) {
        y1 += a2w[(c * 4 + r) * 9 + 4] * ha[r];
        y2 += m2w[c * 4 + r] * hm[r];
    }
    ca[b * C_CH + c] = sigmoidf(sigmoidf(y1) + sigmoidf(y2));
}

// --------------------------------------------------------- adaptive pool ---
__global__ void k_apool(const float* __restrict__ in, float* __restrict__ out,
                        int oh, int ow) {
    int idx = blockIdx.x * blockDim.x + threadIdx.x;
    int total = BATCH * C_CH * oh * ow;
    if (idx >= total) return;
    int ox = idx % ow;
    int t = idx / ow;
    int oy = t % oh;
    int bc = t / oh;
    int sy = oy * HH / oh, ey = ((oy + 1) * HH + oh - 1) / oh;
    int sx = ox * WW / ow, ex = ((ox + 1) * WW + ow - 1) / ow;
    const float* ip = in + (long)bc * NPIX;
    float s = 0.f;
    for (int y = sy; y < ey; ++y)
        for (int x = sx; x < ex; ++x) s += ip[y * WW + x];
    out[idx] = s / (float)((ey - sy) * (ex - sx));
}

__global__ void k_cat_add(const float* __restrict__ pool, const float* __restrict__ h2,
                          float* __restrict__ cat, int sz, int off) {
    int idx = blockIdx.x * blockDim.x + threadIdx.x;
    if (idx >= BATCH * C_CH * sz) return;
    int j = idx % sz, bc = idx / sz;
    cat[(long)bc * LPAD + off + j] = pool[idx] + h2[idx];
}
__global__ void k_zeropad(float* __restrict__ cat) {
    int idx = blockIdx.x * blockDim.x + threadIdx.x;
    int npad = LPAD - LVALID;
    if (idx >= BATCH * C_CH * npad) return;
    int j = idx % npad, bc = idx / npad;
    cat[(long)bc * LPAD + LVALID + j] = 0.f;
}

// ------------------------------------------------- layernorm over C (B,L) ---
__global__ void k_layernorm(float* __restrict__ cat, const float* __restrict__ g,
                            const float* __restrict__ bb) {
    int idx = blockIdx.x * blockDim.x + threadIdx.x;
    if (idx >= BATCH * LVALID) return;
    int l = idx % LVALID, b = idx / LVALID;
    float* base = cat + (long)b * C_CH * LPAD + l;
    float mu = 0.f;
    for (int c = 0; c < C_CH; ++c) mu += base[c * LPAD];
    mu *= (1.0f / C_CH);
    float var = 0.f;
    for (int c = 0; c < C_CH; ++c) { float d = base[c * LPAD] - mu; var += d * d; }
    var *= (1.0f / C_CH);
    float rs = rsqrtf(var + 1e-5f);
    for (int c = 0; c < C_CH; ++c)
        base[c * LPAD] = (base[c * LPAD] - mu) * rs * g[c] + bb[c];
}

// ---------------------------------------------------------------- l2norm ---
__global__ void k_l2norm_q(float* __restrict__ q) {
    int idx = blockIdx.x * blockDim.x + threadIdx.x;
    if (idx >= BATCH * HEADS * NPIX) return;
    int n = idx & (NPIX - 1);
    int t = idx >> 10;
    int h = t & 3, b = t >> 2;
    float* base = q + ((long)b * C_CH + h * DHEAD) * NPIX + n;
    float ss = 0.f;
    for (int j = 0; j < DHEAD; ++j) { float v = base[j * NPIX]; ss += v * v; }
    float inv = 1.0f / fmaxf(sqrtf(ss), 1e-12f);
    for (int j = 0; j < DHEAD; ++j) base[j * NPIX] *= inv;
}
__global__ void k_l2norm_k(float* __restrict__ kv) {
    int idx = blockIdx.x * blockDim.x + threadIdx.x;
    if (idx >= BATCH * HEADS * LPAD) return;
    int l = idx % LPAD;
    int t = idx / LPAD;
    int h = t & 3, b = t >> 2;
    float* base = kv + ((long)b * 2 * C_CH + h * DHEAD) * LPAD + l;
    float ss = 0.f;
    for (int j = 0; j < DHEAD; ++j) { float v = base[j * LPAD]; ss += v * v; }
    float inv = 1.0f / fmaxf(sqrtf(ss), 1e-12f);
    for (int j = 0; j < DHEAD; ++j) base[j * LPAD] *= inv;
}

// ------------------------------------------------------ flash attention ---
// One wave per (b, h, 16-row n-tile). q (B,C,N) normalized; kv (B,2C,Lp)
// with k half normalized. Output attn (B,C,N).
__global__ __launch_bounds__(128)
void k_flash_attn(const float* __restrict__ q, const float* __restrict__ kv,
                  float* __restrict__ outb) {
    __shared__ _Float16 lds[4][16][34];
    int wv = threadIdx.x >> 5, lane = threadIdx.x & 31;
    int hi = lane >> 4, lr = lane & 15;
    int n0 = (blockIdx.x * 4 + wv) * 16;
    int h = blockIdx.y, b = blockIdx.z;
    const float* qb = q + ((long)b * C_CH + h * DHEAD) * NPIX;
    const float* kb = kv + ((long)b * 2 * C_CH + h * DHEAD) * LPAD;
    const float* vb = kv + ((long)b * 2 * C_CH + C_CH + h * DHEAD) * LPAD;

    v16h aq;
#pragma unroll
    for (int j = 0; j < 16; ++j) {
        int k = a_kmap(j >> 1, hi) + (j & 1);
        aq[j] = (_Float16)qb[k * NPIX + n0 + lr];
    }
    v8f acc0 = {0.f, 0.f, 0.f, 0.f, 0.f, 0.f, 0.f, 0.f};
    v8f acc1 = {0.f, 0.f, 0.f, 0.f, 0.f, 0.f, 0.f, 0.f};
    float mrow[8], lrow[8];
#pragma unroll
    for (int r = 0; r < 8; ++r) { mrow[r] = -1e30f; lrow[r] = 0.f; }

    for (int l0 = 0; l0 < LPAD; l0 += 32) {
        if (l0 + 32 < LPAD) {  // prefetch next K/V chunk (global_prefetch_b8)
            __builtin_prefetch(&kb[lr * LPAD + l0 + 32], 0, 1);
            __builtin_prefetch(&vb[lr * LPAD + l0 + 32], 0, 1);
        }
        v16h bk0, bk1;
#pragma unroll
        for (int j = 0; j < 8; ++j) {
            int kd = (hi ? 16 : 0) + 2 * j;
            bk0[2 * j]     = (_Float16)kb[kd * LPAD + l0 + lr];
            bk0[2 * j + 1] = (_Float16)kb[(kd + 1) * LPAD + l0 + lr];
            bk1[2 * j]     = (_Float16)kb[kd * LPAD + l0 + 16 + lr];
            bk1[2 * j + 1] = (_Float16)kb[(kd + 1) * LPAD + l0 + 16 + lr];
        }
        v8f zc = {0.f, 0.f, 0.f, 0.f, 0.f, 0.f, 0.f, 0.f};
        v8f s0 = __builtin_amdgcn_wmma_f32_16x16x32_f16(false, aq, false, bk0,
                                                        (short)0, zc, false, false);
        v8f s1 = __builtin_amdgcn_wmma_f32_16x16x32_f16(false, aq, false, bk1,
                                                        (short)0, zc, false, false);
        float msk0 = (l0 + lr < LVALID) ? 0.f : -1e30f;
        float msk1 = (l0 + 16 + lr < LVALID) ? 0.f : -1e30f;
        float p0[8], p1[8];
#pragma unroll
        for (int r = 0; r < 8; ++r) {
            float a0 = s0[r] * ATT_SCALE + msk0;
            float a1 = s1[r] * ATT_SCALE + msk1;
            float mx = halfmax(fmaxf(a0, a1));
            float mnew = fmaxf(mrow[r], mx);
            float corr = __expf(mrow[r] - mnew);
            p0[r] = __expf(a0 - mnew);
            p1[r] = __expf(a1 - mnew);
            lrow[r] = lrow[r] * corr + halfsum(p0[r] + p1[r]);
            mrow[r] = mnew;
            acc0[r] *= corr;
            acc1[r] *= corr;
        }
        // relayout P: D-layout -> A-fragment, via per-wave LDS tile
#pragma unroll
        for (int r = 0; r < 8; ++r) {
            int row = r + hi * 8;
            lds[wv][row][lr] = (_Float16)p0[r];
            lds[wv][row][16 + lr] = (_Float16)p1[r];
        }
        __builtin_amdgcn_wave_barrier();
        v16h pf;
#pragma unroll
        for (int j = 0; j < 16; ++j) {
            int k = a_kmap(j >> 1, hi) + (j & 1);
            pf[j] = lds[wv][lr][k];
        }
        __builtin_amdgcn_wave_barrier();
        v16h bv0, bv1;
#pragma unroll
        for (int j = 0; j < 8; ++j) {
            int kl = (hi ? 16 : 0) + 2 * j;
            bv0[2 * j]     = (_Float16)vb[lr * LPAD + l0 + kl];
            bv0[2 * j + 1] = (_Float16)vb[lr * LPAD + l0 + kl + 1];
            bv1[2 * j]     = (_Float16)vb[(16 + lr) * LPAD + l0 + kl];
            bv1[2 * j + 1] = (_Float16)vb[(16 + lr) * LPAD + l0 + kl + 1];
        }
        acc0 = __builtin_amdgcn_wmma_f32_16x16x32_f16(false, pf, false, bv0,
                                                      (short)0, acc0, false, false);
        acc1 = __builtin_amdgcn_wmma_f32_16x16x32_f16(false, pf, false, bv1,
                                                      (short)0, acc1, false, false);
    }
#pragma unroll
    for (int r = 0; r < 8; ++r) {
        float inv = (lrow[r] > 0.f) ? 1.0f / lrow[r] : 0.f;
        int n = n0 + r + hi * 8;
        outb[((long)b * C_CH + h * DHEAD + lr) * NPIX + n] = acc0[r] * inv;
        outb[((long)b * C_CH + h * DHEAD + 16 + lr) * NPIX + n] = acc1[r] * inv;
    }
}

// ================================================================ launch ===
extern "C" void kernel_launch(void* const* d_in, const int* in_sizes, int n_in,
                              void* d_out, int out_size, void* d_ws, size_t ws_size,
                              hipStream_t stream) {
    (void)in_sizes; (void)n_in; (void)out_size; (void)ws_size;
    const float* x          = (const float*)d_in[0];
    const float* lka_conv_w = (const float*)d_in[1];
    const float* lka_conv_b = (const float*)d_in[2];
    const float* lka_c0_w   = (const float*)d_in[3];
    const float* lka_c0_b   = (const float*)d_in[4];
    const float* lka_sp_w   = (const float*)d_in[5];
    const float* lka_sp_b   = (const float*)d_in[6];
    const float* lka_1x1_w  = (const float*)d_in[7];
    const float* lka_1x1_b  = (const float*)d_in[8];
    const float* q_w        = (const float*)d_in[9];
    const float* kv_w       = (const float*)d_in[10];
    const float* ca_a1_w    = (const float*)d_in[11];
    const float* ca_a1_b    = (const float*)d_in[12];
    const float* ca_a2_w    = (const float*)d_in[13];
    const float* ca_a2_b    = (const float*)d_in[14];
    const float* ca_m1_w    = (const float*)d_in[15];
    const float* ca_m1_b    = (const float*)d_in[16];
    const float* ca_m2_w    = (const float*)d_in[17];
    const float* ca_m2_b    = (const float*)d_in[18];
    const float* conv3_w    = (const float*)d_in[19];
    const float* conv3_b    = (const float*)d_in[20];
    const float* ln_g       = (const float*)d_in[21];
    const float* ln_b       = (const float*)d_in[22];
    const float* lepe1_w    = (const float*)d_in[23];
    const float* lepe1_b    = (const float*)d_in[24];
    const float* lepe2_w    = (const float*)d_in[25];
    const float* lepe2_b    = (const float*)d_in[26];
    const float* proj_w     = (const float*)d_in[27];
    const float* proj_b     = (const float*)d_in[28];
    const float* dc_h_w[4], *dc_h_b[4], *dc_v_w[4], *dc_v_b[4];
    for (int i = 0; i < 4; ++i) {
        dc_h_w[i] = (const float*)d_in[29 + 4 * i];
        dc_h_b[i] = (const float*)d_in[30 + 4 * i];
        dc_v_w[i] = (const float*)d_in[31 + 4 * i];
        dc_v_b[i] = (const float*)d_in[32 + 4 * i];
    }
    float* out = (float*)d_out;

    const long SZ = (long)BATCH * C_CH * NPIX;
    float* W = (float*)d_ws;
    float* x_img = W;
    float* t0    = W + 1 * SZ;
    float* t1    = W + 2 * SZ;
    float* x1    = W + 3 * SZ;
    float* qbuf  = W + 4 * SZ;
    float* x2b   = W + 5 * SZ;
    float* x3b   = W + 6 * SZ;
    float* poolb = W + 7 * SZ;
    float* hb    = W + 8 * SZ;
    float* h2b   = W + 9 * SZ;
    float* abuf  = W + 10 * SZ;
    float* lt1   = W + 11 * SZ;
    float* lt2   = W + 12 * SZ;
    float* cat   = W + 13 * SZ;                        // B*C*LPAD
    float* kvb   = cat + (long)BATCH * C_CH * LPAD;    // B*2C*LPAD
    float* cab   = kvb + (long)BATCH * 2 * C_CH * LPAD;

    const int TPB = 256;
    const int NB_FULL = (int)((SZ + TPB - 1) / TPB);

    // 1) (B,N,C) -> (B,C,N)
    k_transpose<<<NB_FULL, TPB, 0, stream>>>(x, x_img);
    // 2) LKA depthwise chain: 3x3 -> 5x5 -> 7x7 dil3
    k_dwconv<<<NB_FULL, TPB, 0, stream>>>(x_img, lka_conv_w, lka_conv_b, t0,
                                          HH, WW, 3, 3, 1, 1, 1, 1, 0);
    k_dwconv<<<NB_FULL, TPB, 0, stream>>>(t0, lka_c0_w, lka_c0_b, t1,
                                          HH, WW, 5, 5, 2, 2, 1, 1, 0);
    k_dwconv<<<NB_FULL, TPB, 0, stream>>>(t1, lka_sp_w, lka_sp_b, t0,
                                          HH, WW, 7, 7, 9, 9, 3, 3, 0);
    // 3) LKA 1x1 conv (GEMM) then gate: x1 = x_img * a
    k_gemm_wmma<<<dim3(NPIX / 64, C_CH / 64, BATCH), 128, 0, stream>>>(
        lka_1x1_w, lka_1x1_b, t0, t1, C_CH, NPIX, C_CH,
        (long)C_CH * NPIX, NPIX, (long)C_CH * NPIX, 0);
    k_mul<<<NB_FULL, TPB, 0, stream>>>(x_img, t1, x1, (int)SZ);
    // 4) q projection + cosine norm
    k_gemm_wmma<<<dim3(NPIX / 64, C_CH / 64, BATCH), 128, 0, stream>>>(
        q_w, nullptr, x1, qbuf, C_CH, NPIX, C_CH,
        (long)C_CH * NPIX, NPIX, (long)C_CH * NPIX, 0);
    k_l2norm_q<<<(BATCH * HEADS * NPIX + TPB - 1) / TPB, TPB, 0, stream>>>(qbuf);
    // 5) channel attention gate -> x2
    k_chanatt<<<BATCH, C_CH, 0, stream>>>(x_img, ca_a1_w, ca_a1_b, ca_a2_w, ca_a2_b,
                                          ca_m1_w, ca_m1_b, ca_m2_w, ca_m2_b, cab);
    k_gate_ca<<<NB_FULL, TPB, 0, stream>>>(x_img, cab, x2b);
    // 6) deformable-conv (zero-offset) = dense 3x3 conv, implicit GEMM
    k_conv3_wmma<<<dim3(NPIX / 16, C_CH / 16, BATCH), 32, 0, stream>>>(
        conv3_w, conv3_b, x2b, x3b);
    // 7) multi-scale pooled KV path
    const int psz[4] = {32, 16, 11, 5};
    const int off[4] = {0, 1024, 1280, 1401};
    for (int i = 0; i < 4; ++i) {
        int oh = psz[i], sz = oh * oh;
        int tot = BATCH * C_CH * sz;
        int nb = (tot + TPB - 1) / TPB;
        k_apool<<<nb, TPB, 0, stream>>>(x3b, poolb, oh, oh);
        k_dwconv<<<nb, TPB, 0, stream>>>(poolb, dc_h_w[i], dc_h_b[i], hb,
                                         oh, oh, 1, 7, 0, 9, 1, 3, 0);
        k_dwconv<<<nb, TPB, 0, stream>>>(hb, dc_v_w[i], dc_v_b[i], h2b,
                                         oh, oh, 7, 1, 9, 0, 3, 1, 1);
        k_cat_add<<<nb, TPB, 0, stream>>>(poolb, h2b, cat, sz, off[i]);
    }
    k_zeropad<<<(BATCH * C_CH * (LPAD - LVALID) + TPB - 1) / TPB, TPB, 0, stream>>>(cat);
    // 8) layernorm over C (in place), kv projection, k cosine norm
    k_layernorm<<<(BATCH * LVALID + TPB - 1) / TPB, TPB, 0, stream>>>(cat, ln_g, ln_b);
    k_gemm_wmma<<<dim3(LPAD / 64, 2 * C_CH / 64, BATCH), 128, 0, stream>>>(
        kv_w, nullptr, cat, kvb, 2 * C_CH, LPAD, C_CH,
        (long)C_CH * LPAD, LPAD, (long)2 * C_CH * LPAD, 0);
    k_l2norm_k<<<(BATCH * HEADS * LPAD + TPB - 1) / TPB, TPB, 0, stream>>>(kvb);
    // 9) flash attention -> abuf (B,C,N)
    k_flash_attn<<<dim3(NPIX / 64, HEADS, BATCH), 128, 0, stream>>>(qbuf, kvb, abuf);
    // 10) LEPE on LKA output x1, add into abuf
    k_dwconv<<<NB_FULL, TPB, 0, stream>>>(x1, lepe1_w, lepe1_b, lt1,
                                          HH, WW, 1, 3, 0, 1, 1, 1, 0);
    k_dwconv<<<NB_FULL, TPB, 0, stream>>>(lt1, lepe2_w, lepe2_b, lt2,
                                          HH, WW, 3, 1, 1, 0, 1, 1, 1);
    k_add_inplace<<<NB_FULL, TPB, 0, stream>>>(abuf, lt2, (int)SZ);
    // 11) output projection, written as (B,N,C)
    k_gemm_wmma<<<dim3(NPIX / 64, C_CH / 64, BATCH), 128, 0, stream>>>(
        proj_w, proj_b, abuf, out, C_CH, NPIX, C_CH,
        (long)C_CH * NPIX, NPIX, (long)NPIX * C_CH, 1);
}
